// Attention_89867895702157
// MI455X (gfx1250) — matrix operand order repaired and checked by reading, not compiled, not run on hardware
//
#include <hip/hip_runtime.h>

typedef _Float16 half_t;
typedef __attribute__((ext_vector_type(16))) _Float16 v16h;
typedef __attribute__((ext_vector_type(8)))  _Float16 v8h;
typedef __attribute__((ext_vector_type(4)))  _Float16 v4h;
typedef __attribute__((ext_vector_type(8)))  float    v8f;

#define B_   128
#define N_   197
#define NP   224            // padded token count, 7*32
#define H_   12
#define DH   64
#define DIM  768
#define M_   (B_ * N_)      // 25216 = 197 * 128
#define K3   (3 * DIM)      // 2304

static constexpr float SCALE = 0.125f;  // DH^-0.5

static __device__ inline v8f wmma_f16(v16h a, v16h b, v8f c) {
  // D = A(16x32 f16) * B(32x16 f16) + C(16x16 f32)
  return __builtin_amdgcn_wmma_f32_16x16x32_f16(
      /*neg_a=*/false, a, /*neg_b=*/false, b,
      /*c_mod=*/(short)0, c, /*reuse_a=*/false, /*reuse_b=*/false);
}

// Async global->LDS B128 copy (ASYNCcnt tracked). lds_off = per-lane LDS byte
// offset (low 32 bits of the flat shared address), gaddr = per-lane global addr.
static __device__ inline void async_g2l_b128(unsigned lds_off, const void* gaddr) {
  asm volatile("global_load_async_to_lds_b128 %0, %1, off"
               :: "v"(lds_off), "v"((unsigned long long)(uintptr_t)gaddr)
               : "memory");
}
static __device__ inline void wait_async0() {
  asm volatile("s_wait_asynccnt 0" ::: "memory");
}
static __device__ inline unsigned lds_off32(const void* p) {
  return (unsigned)(uintptr_t)p;   // flat LDS addr = {aperture_hi, offset_lo}
}

// ---------------------------------------------------------------------------
// Kernel 0: one-time f32 -> f16 conversion (x, qkv_w, proj_w). Moves all
// conversion VALU out of the GEMM mainloop and halves re-streamed A traffic.
// ---------------------------------------------------------------------------
__global__ __launch_bounds__(256) void cvt_f32_f16(
    const float* __restrict__ src, half_t* __restrict__ dst, int n4) {
  const int i = blockIdx.x * 256 + threadIdx.x;
  if (i >= n4) return;
  float4 f = *(const float4*)(src + (size_t)i * 4);
  v4h h;
  h.x = (half_t)f.x; h.y = (half_t)f.y; h.z = (half_t)f.z; h.w = (half_t)f.w;
  *(v4h*)(dst + (size_t)i * 4) = h;
}

// ---------------------------------------------------------------------------
// Kernel 1: gather relative-position bias into padded [H, NP, NP] f32 layout.
// Padded key columns get -1e30 so softmax assigns them zero weight.
// ---------------------------------------------------------------------------
__global__ __launch_bounds__(256) void bias_gather(
    const float* __restrict__ table, const int* __restrict__ ridx,
    float* __restrict__ bias) {
  const long long i = (long long)blockIdx.x * blockDim.x + threadIdx.x;
  const long long total = (long long)H_ * NP * NP;
  if (i >= total) return;
  const int n = (int)(i % NP);
  const long long t2 = i / NP;
  const int m = (int)(t2 % NP);
  const int h = (int)(t2 / NP);
  float v = -1.0e30f;
  if (m < N_ && n < N_) v = table[(size_t)ridx[m * N_ + n] * H_ + h];
  bias[i] = v;
}

// ---------------------------------------------------------------------------
// Kernel 2: zero pad rows of q/k (rows 197..223) and pad cols of v^T.
// ---------------------------------------------------------------------------
__global__ __launch_bounds__(256) void pad_zero(
    half_t* __restrict__ qb, half_t* __restrict__ kb, half_t* __restrict__ vTb) {
  const int i = blockIdx.x * blockDim.x + threadIdx.x;
  const int total = B_ * H_ * (NP - N_) * DH;
  if (i >= total) return;
  const int d = i & (DH - 1);
  const int rest = i >> 6;
  const int row = N_ + rest % (NP - N_);
  const int bh = rest / (NP - N_);
  qb[((size_t)bh * NP + row) * DH + d]  = (half_t)0.f;
  kb[((size_t)bh * NP + row) * DH + d]  = (half_t)0.f;
  vTb[((size_t)bh * DH + d) * NP + row] = (half_t)0.f;
}

// ---------------------------------------------------------------------------
// WMMA GEMM:  C[M x Nout] = A[M x K] * W[Nout x K]^T, A and W in f16.
// Block tile 128x128, 256 threads = 8 waves; wave tile 32x64 (2x4 frags).
// Double-buffered LDS filled by async global->LDS B128 DMA; the next tile's
// DMA overlaps the current tile's 8 WMMAs. ASYNCcnt + barrier per K-step.
// EPI 0: QKV epilogue -> scatter to q (scaled), k, v^T (f16)
// EPI 1: proj epilogue -> +bias, f32 out
// ---------------------------------------------------------------------------
template <int EPI>
__global__ __launch_bounds__(256) void gemm_wmma(
    const half_t* __restrict__ Ap, const half_t* __restrict__ Wp,
    const float* __restrict__ bias,
    half_t* __restrict__ qb, half_t* __restrict__ kb, half_t* __restrict__ vTb,
    float* __restrict__ Cout, int K) {
  __shared__ __attribute__((aligned(16))) half_t As[2][128][40];  // 32 K + pad
  __shared__ __attribute__((aligned(16))) half_t Bs[2][128][40];

  const int t    = threadIdx.x;
  const int lane = t & 31;
  const int wave = t >> 5;
  const int l16  = lane & 15;
  const int lh   = lane >> 4;
  const int wm   = wave >> 1;      // 0..3 -> M offset wm*32
  const int wn   = wave & 1;       // 0..1 -> N offset wn*64

  const int n_blk = blockIdx.x * 128;
  const int m_blk = blockIdx.y * 128;

  // Async-copy chunk mapping: tile = 128 rows x 64B = 512 B128 chunks,
  // 2 chunks per thread. row = c>>2, 16B segment = c&3.
  const int c0 = t * 2;

  auto issue = [&](int k0, int buf) {
#pragma unroll
    for (int j = 0; j < 2; ++j) {
      const int c = c0 + j;
      const int row = c >> 2;
      const int seg = (c & 3) * 16;                       // byte segment
      async_g2l_b128(lds_off32(&As[buf][row][0]) + seg,
                     (const char*)(Ap + (size_t)(m_blk + row) * K + k0) + seg);
      async_g2l_b128(lds_off32(&Bs[buf][row][0]) + seg,
                     (const char*)(Wp + (size_t)(n_blk + row) * K + k0) + seg);
    }
  };

  v8f acc[2][4] = {};
  const int nk = K / 32;

  issue(0, 0);
  for (int kt = 0; kt < nk; ++kt) {
    wait_async0();                 // this wave's tile kt is in LDS
    __syncthreads();               // everyone's tile kt is in LDS
    if (kt + 1 < nk) issue((kt + 1) * 32, (kt + 1) & 1);  // overlap with WMMA
    const int buf = kt & 1;

    v16h af[2];
#pragma unroll
    for (int mf = 0; mf < 2; ++mf) {
      const half_t* s = &As[buf][wm * 32 + mf * 16 + l16][lh * 8];
      ((v8h*)&af[mf])[0] = *(const v8h*)s;          // k = lh*8 .. +8
      ((v8h*)&af[mf])[1] = *(const v8h*)(s + 16);   // k = 16+lh*8 .. +8
    }
    v16h bf[4];
#pragma unroll
    for (int nf = 0; nf < 4; ++nf) {
      const half_t* s = &Bs[buf][wn * 64 + nf * 16 + l16][lh * 8];
      ((v8h*)&bf[nf])[0] = *(const v8h*)s;          // k = lh*8 .. +8
      ((v8h*)&bf[nf])[1] = *(const v8h*)(s + 16);
    }
#pragma unroll
    for (int mf = 0; mf < 2; ++mf)
#pragma unroll
      for (int nf = 0; nf < 4; ++nf)
        acc[mf][nf] = wmma_f16(af[mf], bf[nf], acc[mf][nf]);
  }

  // ---- epilogue: C layout is VGPR r <-> M = r + lh*8, lane&15 <-> N ----
#pragma unroll
  for (int mf = 0; mf < 2; ++mf) {
#pragma unroll
    for (int nf = 0; nf < 4; ++nf) {
      const int n = n_blk + wn * 64 + nf * 16 + l16;
      const int mbase = m_blk + wm * 32 + mf * 16 + lh * 8;
#pragma unroll
      for (int r = 0; r < 8; ++r) {
        const int m = mbase + r;
        const float val = acc[mf][nf][r];
        if constexpr (EPI == 0) {
          const int which = n / DIM;          // 0=q 1=k 2=v
          const int rem = n - which * DIM;
          const int h = rem >> 6;
          const int d = rem & 63;
          const int b = m / N_;
          const int nn = m - b * N_;
          const size_t bh = (size_t)b * H_ + h;
          if (which == 0)      qb[(bh * NP + nn) * DH + d]  = (half_t)(val * SCALE);
          else if (which == 1) kb[(bh * NP + nn) * DH + d]  = (half_t)val;
          else                 vTb[(bh * DH + d) * NP + nn] = (half_t)val;
        } else {
          Cout[(size_t)m * DIM + n] = val + bias[n];
        }
      }
    }
  }
}

// ---------------------------------------------------------------------------
// Kernel 4: attention. One wave per (16 query rows, h, b).
// S = q*k^T (14 reg fragments over 224 padded keys) + bias, softmax with
// shfl_xor row reductions, P staged via LDS (C->A layout), then P*v^T.
// ---------------------------------------------------------------------------
__global__ __launch_bounds__(32) void attn_softmax_pv(
    const half_t* __restrict__ qb, const half_t* __restrict__ kb,
    const half_t* __restrict__ vTb, const float* __restrict__ bias,
    half_t* __restrict__ aout) {
  __shared__ __attribute__((aligned(16))) half_t Ps[16][40];

  const int mf = blockIdx.x;        // 0..12 (rows 0..207; only m<197 stored)
  const int h  = blockIdx.y;
  const int b  = blockIdx.z;
  const int lane = threadIdx.x;
  const int l16 = lane & 15;
  const int lh  = lane >> 4;
  const int m0  = mf * 16;

  const size_t bh = (size_t)b * H_ + h;
  const half_t* qp = qb  + bh * (size_t)NP * DH;
  const half_t* kp = kb  + bh * (size_t)NP * DH;
  const half_t* vp = vTb + bh * (size_t)DH * NP;

  // q A-fragments for the two K=32 steps over DH=64
  v16h qf[2];
#pragma unroll
  for (int ks = 0; ks < 2; ++ks) {
    const half_t* s = qp + (size_t)(m0 + l16) * DH + ks * 32 + lh * 8;
    ((v8h*)&qf[ks])[0] = *(const v8h*)s;
    ((v8h*)&qf[ks])[1] = *(const v8h*)(s + 16);
  }

  // S = q * k^T : 14 column fragments of 16 keys each
  v8f acc[14];
#pragma unroll
  for (int nf = 0; nf < 14; ++nf) {
    v8f c = {};
#pragma unroll
    for (int ks = 0; ks < 2; ++ks) {
      v16h kf;
      const half_t* s = kp + (size_t)(nf * 16 + l16) * DH + ks * 32 + lh * 8;
      ((v8h*)&kf)[0] = *(const v8h*)s;
      ((v8h*)&kf)[1] = *(const v8h*)(s + 16);
      c = wmma_f16(qf[ks], kf, c);
    }
    acc[nf] = c;
  }

  // bias + softmax (row m = m0 + lh*8 + r, col n = nf*16 + l16)
  const float* bp = bias + ((size_t)h * NP + (m0 + lh * 8)) * NP;
  float rmax[8], rsum[8];
#pragma unroll
  for (int r = 0; r < 8; ++r) rmax[r] = -3.0e38f;
#pragma unroll
  for (int nf = 0; nf < 14; ++nf)
#pragma unroll
    for (int r = 0; r < 8; ++r) {
      float a = acc[nf][r] + bp[(size_t)r * NP + nf * 16 + l16];
      acc[nf][r] = a;
      rmax[r] = fmaxf(rmax[r], a);
    }
#pragma unroll
  for (int r = 0; r < 8; ++r) {
#pragma unroll
    for (int off = 1; off < 16; off <<= 1)   // reduce within 16-lane half
      rmax[r] = fmaxf(rmax[r], __shfl_xor(rmax[r], off, 32));
    rsum[r] = 0.0f;
  }
#pragma unroll
  for (int nf = 0; nf < 14; ++nf)
#pragma unroll
    for (int r = 0; r < 8; ++r) {
      float e = __expf(acc[nf][r] - rmax[r]);
      acc[nf][r] = e;
      rsum[r] += e;
    }
#pragma unroll
  for (int r = 0; r < 8; ++r) {
#pragma unroll
    for (int off = 1; off < 16; off <<= 1)
      rsum[r] += __shfl_xor(rsum[r], off, 32);
    rsum[r] = 1.0f / rsum[r];
  }

  // out = P * V : contraction over 224 keys in 7 steps of 32
  v8f oacc[4] = {};
  for (int ns = 0; ns < 7; ++ns) {
    __syncthreads();
    // stage P (C layout -> LDS), normalized + f16
#pragma unroll
    for (int j = 0; j < 2; ++j) {
      const int nf = ns * 2 + j;
#pragma unroll
      for (int r = 0; r < 8; ++r)
        Ps[lh * 8 + r][j * 16 + l16] = (half_t)(acc[nf][r] * rsum[r]);
    }
    __syncthreads();
    v16h pf;   // re-read as A fragment
    ((v8h*)&pf)[0] = *(const v8h*)&Ps[l16][lh * 8];
    ((v8h*)&pf)[1] = *(const v8h*)&Ps[l16][16 + lh * 8];
#pragma unroll
    for (int df = 0; df < 4; ++df) {
      v16h vf;  // B[kk][d] = vT[d][kk], contiguous in kk
      const half_t* s = vp + (size_t)(df * 16 + l16) * NP + ns * 32 + lh * 8;
      ((v8h*)&vf)[0] = *(const v8h*)s;
      ((v8h*)&vf)[1] = *(const v8h*)(s + 16);
      oacc[df] = wmma_f16(pf, vf, oacc[df]);
    }
  }

  // store [b, m, h*64+d] as f16 (proj GEMM input), guard padded query rows
#pragma unroll
  for (int df = 0; df < 4; ++df) {
    const int d = h * DH + df * 16 + l16;
#pragma unroll
    for (int r = 0; r < 8; ++r) {
      const int m = m0 + lh * 8 + r;
      if (m < N_)
        aout[((size_t)b * N_ + m) * DIM + d] = (half_t)oacc[df][r];
    }
  }
}

// ---------------------------------------------------------------------------
extern "C" void kernel_launch(void* const* d_in, const int* in_sizes, int n_in,
                              void* d_out, int out_size, void* d_ws, size_t ws_size,
                              hipStream_t stream) {
  const float* x      = (const float*)d_in[0];
  const float* qkv_w  = (const float*)d_in[1];
  const float* proj_w = (const float*)d_in[2];
  const float* proj_b = (const float*)d_in[3];
  const float* table  = (const float*)d_in[4];
  const int*   ridx   = (const int*)d_in[5];
  float* out = (float*)d_out;

  // workspace carve-up (~217 MB)
  const size_t qkvElems  = (size_t)B_ * H_ * NP * DH;   // 22,020,096 halves each
  const size_t xElems    = (size_t)M_ * DIM;            // 19,365,888
  const size_t wqkvElems = (size_t)K3 * DIM;            //  1,769,472
  const size_t wpjElems  = (size_t)DIM * DIM;           //    589,824
  half_t* qb    = (half_t*)d_ws;
  half_t* kb    = qb + qkvElems;
  half_t* vTb   = kb + qkvElems;
  float*  bias  = (float*)(vTb + qkvElems);
  half_t* attn  = (half_t*)(bias + (size_t)H_ * NP * NP);
  half_t* xh    = attn + xElems;
  half_t* wqkvh = xh + xElems;
  half_t* wpjh  = wqkvh + wqkvElems;

  // one-time f32 -> f16 conversions
  cvt_f32_f16<<<(unsigned)((xElems / 4 + 255) / 256), 256, 0, stream>>>(
      x, xh, (int)(xElems / 4));
  cvt_f32_f16<<<(unsigned)((wqkvElems / 4 + 255) / 256), 256, 0, stream>>>(
      qkv_w, wqkvh, (int)(wqkvElems / 4));
  cvt_f32_f16<<<(unsigned)((wpjElems / 4 + 255) / 256), 256, 0, stream>>>(
      proj_w, wpjh, (int)(wpjElems / 4));

  {
    const long long tot = (long long)H_ * NP * NP;
    bias_gather<<<(unsigned)((tot + 255) / 256), 256, 0, stream>>>(table, ridx, bias);
  }
  {
    const int tot = B_ * H_ * (NP - N_) * DH;
    pad_zero<<<(tot + 255) / 256, 256, 0, stream>>>(qb, kb, vTb);
  }
  // QKV: [25216 x 768] @ [2304 x 768]^T -> scatter q/k/v^T
  gemm_wmma<0><<<dim3(K3 / 128, M_ / 128), 256, 0, stream>>>(
      xh, wqkvh, nullptr, qb, kb, vTb, nullptr, DIM);
  // attention
  attn_softmax_pv<<<dim3(13, H_, B_), 32, 0, stream>>>(qb, kb, vTb, bias, attn);
  // proj: [25216 x 768] @ [768 x 768]^T + b -> f32 out
  gemm_wmma<1><<<dim3(DIM / 128, M_ / 128), 256, 0, stream>>>(
      attn, wpjh, proj_b, nullptr, nullptr, nullptr, out, DIM);
}
